// ImgExemplarSelfAttn_43052752175736
// MI455X (gfx1250) — matrix operand order repaired and checked by reading, not compiled, not run on hardware
//
#include <hip/hip_runtime.h>

// ---------------------------------------------------------------------------
// ImgExemplarSelfAttn for MI455X (gfx1250, wave32, WMMA, async-LDS if avail).
// B=4, C=256, HW=4096, N=4097, NQ=150.
//   K1  T = U^T * sim (fp32, exact -> top-k) ; also emits T16 (bf16) and
//       Cc16[b][m][d] (bf16 transposed img) for the attention kernel.
//   K2  V = Cc^T * v_w^T (fp32) ; also emits V16[b][d][m] (bf16 transposed).
//   K3  scores = exe column of attn (fp32 exact).
//   K4  NMS + iterative top-150.
//   K5  fused flash attention: A-fragments hoisted to registers (global b128
//       loads), chunk staging via async global->LDS b128 when available,
//       bf16 WMMA for QK and PV, fp32 online softmax, out1 = O + img.
//   K6  gather queries.
// ---------------------------------------------------------------------------

#define HW      4096
#define CDIM    256
#define NFULL   4097
#define NQ      150
#define NEGV    1000000000.0f
#define NEGINF  (-3.0e38f)

#define BCPITCH 264   // Cc chunk LDS pitch [m][d] (256 + 8), 16B-aligned rows
#define BVPITCH 40    // V chunk LDS pitch [d][m] (32 + 8)
#define PPITCH  40    // P tile LDS pitch (32 + 8)

typedef __attribute__((ext_vector_type(16))) __bf16 v16bf;
typedef __attribute__((ext_vector_type(8)))  float  v8f;
typedef __attribute__((ext_vector_type(4)))  unsigned int u32x4;
typedef __attribute__((ext_vector_type(4)))  int v4i;

#if defined(__has_builtin)
#if __has_builtin(__builtin_amdgcn_global_load_async_to_lds_b128) && \
    __has_builtin(__builtin_amdgcn_s_wait_asynccnt)
#define HAVE_ASYNC_LDS 1
#endif
#endif

#ifdef HAVE_ASYNC_LDS
// diagnostics from round 2 show the builtin takes pointers to int4
typedef __attribute__((address_space(1))) v4i* gv4i_p;   // global src
typedef __attribute__((address_space(3))) v4i* lv4i_p;   // LDS dst
#endif

__device__ __forceinline__ unsigned short f2bf(float x) {
    unsigned int u = __float_as_uint(x);
    unsigned int r = u + 0x7FFFu + ((u >> 16) & 1u);   // RNE
    return (unsigned short)(r >> 16);
}

// copy 16B global(bf16) -> LDS; async path uses ASYNCcnt, fallback is ld/st.
__device__ __forceinline__ void copy16(const unsigned short* g, unsigned short* l) {
#ifdef HAVE_ASYNC_LDS
    __builtin_amdgcn_global_load_async_to_lds_b128((gv4i_p)g, (lv4i_p)l, 0, 0);
#else
    *reinterpret_cast<u32x4*>(l) = *reinterpret_cast<const u32x4*>(g);
#endif
}
__device__ __forceinline__ void copy_fence() {
#ifdef HAVE_ASYNC_LDS
    __builtin_amdgcn_s_wait_asynccnt(0);
#endif
}

// A-matrix fragment (16x32 bf16), row-major source (global or LDS):
// half0 -> e0..15=K0..15 ; half1 -> K16..31 (contiguous, matches loadBfrag).
__device__ __forceinline__ v16bf loadAfrag(const unsigned short* row, int kBase, int half) {
    union { v16bf v; u32x4 q[2]; } f;
    f.q[0] = *reinterpret_cast<const u32x4*>(row + kBase + 16 * half);
    f.q[1] = *reinterpret_cast<const u32x4*>(row + kBase + 16 * half + 8);
    return f.v;
}
// B-matrix fragment (32x16 bf16): lane column n=l16, contiguous K=16*half..+15.
__device__ __forceinline__ v16bf loadBfrag(const unsigned short* p) {
    union { v16bf v; u32x4 q[2]; } f;
    f.q[0] = *reinterpret_cast<const u32x4*>(p);
    f.q[1] = *reinterpret_cast<const u32x4*>(p + 8);
    return f.v;
}

// ---------------- K1: T = U^T * sim (fp32) + T16 + Cc16 ---------------------
__global__ __launch_bounds__(256) void t_gemm_kernel(
    const float* __restrict__ img, const float* __restrict__ sim,
    float* __restrict__ Tw, unsigned short* __restrict__ T16,
    unsigned short* __restrict__ Cc16)
{
    const int b  = blockIdx.x >> 6;
    const int n0 = (blockIdx.x & 63) * 64;
    const int t  = threadIdx.x;                 // t == d
    __shared__ float sImg[128 * 64];

    float acc[64];
#pragma unroll
    for (int n = 0; n < 64; ++n) acc[n] = 0.0f;

    for (int cp = 0; cp < 2; ++cp) {
        __syncthreads();
        {
            const int c  = cp * 128 + (t >> 1);
            const int x0 = (t & 1) * 32;
            const float* src = img + ((size_t)(b * CDIM + c)) * HW + n0 + x0;
            float* dst = &sImg[(t >> 1) * 64 + x0];
#pragma unroll
            for (int i = 0; i < 32; i += 4) {
                float4 v = *reinterpret_cast<const float4*>(src + i);
                dst[i + 0] = v.x; dst[i + 1] = v.y; dst[i + 2] = v.z; dst[i + 3] = v.w;
            }
        }
        __syncthreads();
        // emit transposed bf16 img tile: Cc16[b][m][c]
        {
            const int n  = t & 63;
            const int c0 = (t >> 6) * 32;
            unsigned short* dst = Cc16 + ((size_t)(b * HW + n0 + n)) * CDIM + cp * 128 + c0;
#pragma unroll
            for (int i = 0; i < 32; ++i) dst[i] = f2bf(sImg[(c0 + i) * 64 + n]);
        }
        for (int c = 0; c < 128; ++c) {
            const float s = sim[(size_t)(cp * 128 + c) * CDIM + t];
            const float* r = &sImg[c * 64];
#pragma unroll
            for (int n = 0; n < 64; ++n) acc[n] = fmaf(r[n], s, acc[n]);
        }
    }
#pragma unroll
    for (int n = 0; n < 64; ++n) {
        const size_t idx = ((size_t)(b * HW + n0 + n)) * CDIM + t;
        Tw[idx]  = acc[n];
        T16[idx] = f2bf(acc[n]);
    }
}

// ---------------- K2: V = Cc^T * v_w^T (fp32) + V16 -------------------------
__global__ __launch_bounds__(256) void v_gemm_kernel(
    const float* __restrict__ img, const float* __restrict__ exe,
    const float* __restrict__ vw, float* __restrict__ Vw,
    unsigned short* __restrict__ V16)
{
    const int b    = blockIdx.x / 65;
    const int tile = blockIdx.x % 65;
    const int t    = threadIdx.x;               // t == d
    __shared__ float sImg[128 * 64];
    __shared__ float sE[256];

    if (tile == 64) {                           // exe (m = 4096) column
        sE[t] = exe[b * CDIM + t] * 1.2f;
        __syncthreads();
        float a = 0.0f;
        for (int c = 0; c < CDIM; ++c) a = fmaf(sE[c], vw[(size_t)t * CDIM + c], a);
        Vw[((size_t)b * NFULL + HW) * CDIM + t] = a;
        return;
    }

    const int n0 = tile * 64;
    float acc[64];
#pragma unroll
    for (int n = 0; n < 64; ++n) acc[n] = 0.0f;

    for (int cp = 0; cp < 2; ++cp) {
        __syncthreads();
        {
            const int c  = cp * 128 + (t >> 1);
            const int x0 = (t & 1) * 32;
            const float* src = img + ((size_t)(b * CDIM + c)) * HW + n0 + x0;
            float* dst = &sImg[(t >> 1) * 64 + x0];
#pragma unroll
            for (int i = 0; i < 32; i += 4) {
                float4 v = *reinterpret_cast<const float4*>(src + i);
                dst[i + 0] = v.x; dst[i + 1] = v.y; dst[i + 2] = v.z; dst[i + 3] = v.w;
            }
        }
        __syncthreads();
        for (int c = 0; c < 128; ++c) {
            const float w = vw[(size_t)t * CDIM + cp * 128 + c];
            const float* r = &sImg[c * 64];
#pragma unroll
            for (int n = 0; n < 64; ++n) acc[n] = fmaf(r[n], w, acc[n]);
        }
    }
#pragma unroll
    for (int n = 0; n < 64; ++n) {
        Vw[((size_t)(b * NFULL + n0 + n)) * CDIM + t] = acc[n];
        V16[((size_t)(b * CDIM + t)) * HW + n0 + n]   = f2bf(acc[n]); // [d][m]
    }
}

// --------------- K3: scores[b,n] = T[b,n,:] . (1.2*exe[b,:]) ----------------
__global__ __launch_bounds__(256) void score_kernel(
    const float* __restrict__ Tw, const float* __restrict__ exe,
    float* __restrict__ scores)
{
    const int b  = blockIdx.x >> 4;
    const int n0 = (blockIdx.x & 15) * 256;
    const int t  = threadIdx.x;
    __shared__ float sE[256];
    sE[t] = exe[b * CDIM + t] * 1.2f;
    __syncthreads();
    const float* r = Tw + ((size_t)(b * HW + n0 + t)) * CDIM;
    float a = 0.0f;
    for (int d = 0; d < CDIM; ++d) a = fmaf(r[d], sE[d], a);
    scores[b * HW + n0 + t] = a;
}

// ------------------- K4: NMS penalty + iterative top-150 --------------------
__global__ __launch_bounds__(256) void topk_kernel(
    const float* __restrict__ scores, int* __restrict__ qids)
{
    const int b = blockIdx.x;
    const int t = threadIdx.x;
    __shared__ float sS[HW];
    __shared__ float rV[256];
    __shared__ int   rI[256];
    const float* sc = scores + (size_t)b * HW;

    for (int k = 0; k < 16; ++k) {
        const int n = t + k * 256;
        const int y = n >> 6, x = n & 63;
        const float c0 = sc[n];
        float pen = 0.0f;
        if (y >= 1 && y <= 62 && x >= 1 && x <= 62) {
            const float up = sc[n - 64], dn = sc[n + 64];
            const float lf = sc[n - 1],  rg = sc[n + 1];
            const bool ismax = (c0 > up) && (c0 >= dn) && (c0 > lf) && (c0 >= rg);
            pen = ismax ? 0.0f : NEGV;
        }
        sS[n] = c0 - pen;
    }
    __syncthreads();

    for (int it = 0; it < NQ; ++it) {
        float bv = NEGINF; int bi = 0x7fffffff;
        for (int k = 0; k < 16; ++k) {
            const int n = t + k * 256;
            const float v = sS[n];
            if (v > bv) { bv = v; bi = n; }       // ascending scan -> lowest idx on tie
        }
        rV[t] = bv; rI[t] = bi;
        __syncthreads();
        for (int s = 128; s > 0; s >>= 1) {
            if (t < s) {
                if (rV[t + s] > rV[t] || (rV[t + s] == rV[t] && rI[t + s] < rI[t])) {
                    rV[t] = rV[t + s]; rI[t] = rI[t + s];
                }
            }
            __syncthreads();
        }
        if (t == 0) { qids[b * NQ + it] = rI[0]; sS[rI[0]] = NEGINF; }
        __syncthreads();
    }
}

// -------- K5: fused flash attention (bf16 WMMA, async-LDS staging) ----------
__global__ __launch_bounds__(256) void attn_kernel(
    const float* __restrict__ img,   const float* __restrict__ mask,
    const unsigned short* __restrict__ T16,
    const unsigned short* __restrict__ Cc16,
    const unsigned short* __restrict__ V16,
    const float* __restrict__ Vw,    const float* __restrict__ scores,
    float* __restrict__ out1)
{
    const int blk = blockIdx.x;
    const int b   = blk >> 6;
    const int n0  = (blk & 63) * 64;
    const int t    = threadIdx.x;
    const int lane = t & 31, wave = t >> 5;
    const int half = lane >> 4, l16 = lane & 15;
    const int rt = wave & 3;      // row tile (16 rows each)
    const int cg = wave >> 2;     // column group

    __shared__ unsigned short sB[CDIM * BVPITCH];   // Cc chunk / V chunk (reused)
    __shared__ unsigned short sP[64 * PPITCH];      // softmax probs, bf16
    __shared__ float sMaskN[64], sMaskM[32];
    __shared__ float sM[64], sL[64], sScale[64], sPexe[64], sExe[64];
    __shared__ float sCMax[128], sCSum[128];

    // ---- A fragments (Q) hoisted for the whole kernel, direct global b128 ----
    const unsigned short* aRow =
        T16 + ((size_t)(b * HW + n0 + rt * 16 + l16)) * CDIM;
    v16bf afr[8];
#pragma unroll
    for (int kk = 0; kk < 8; ++kk) afr[kk] = loadAfrag(aRow, kk * 32, half);

    if (t < 64) {
        sMaskN[t] = mask[b * HW + n0 + t];
        sExe[t]   = scores[b * HW + n0 + t];
        sM[t] = NEGINF;
        sL[t] = 0.0f;
    }
    v8f acc[8];
#pragma unroll
    for (int j = 0; j < 8; ++j) acc[j] = (v8f){0,0,0,0,0,0,0,0};
    __syncthreads();

    for (int mc = 0; mc < 128; ++mc) {
        const int m0 = mc * 32;

        // ---- stage Cc chunk (bf16, pre-transposed): 32 rows x 512B ----
        {
            const unsigned short* base = Cc16 + ((size_t)(b * HW + m0)) * CDIM;
            if (mc < 127) __builtin_prefetch(base + 32 * CDIM, 0, 0);
#pragma unroll
            for (int k2 = 0; k2 < 4; ++k2) {
                const int seg = t + k2 * 256;
                const int mi  = seg >> 5;
                const int off = (seg & 31) * 8;
                copy16(base + (size_t)mi * CDIM + off, &sB[mi * BCPITCH + off]);
            }
        }
        if (t < 32) sMaskM[t] = mask[b * HW + m0 + t];
        copy_fence();
        __syncthreads();

        // ---- QK logits: preload 8 B fragments, then WMMA burst (2 chains) ----
        v8f accL;
        {
            const unsigned short* bRow = sB + (cg * 16 + l16) * BCPITCH;
            v16bf bfr[8];
#pragma unroll
            for (int kk = 0; kk < 8; ++kk)
                bfr[kk] = loadBfrag(bRow + kk * 32 + 16 * half);
            v8f a0 = (v8f){0,0,0,0,0,0,0,0};
            v8f a1 = (v8f){0,0,0,0,0,0,0,0};
#pragma unroll
            for (int kk = 0; kk < 8; kk += 2) {
                a0 = __builtin_amdgcn_wmma_f32_16x16x32_bf16(
                    false, afr[kk],     false, bfr[kk],     (short)0, a0, false, false);
                a1 = __builtin_amdgcn_wmma_f32_16x16x32_bf16(
                    false, afr[kk + 1], false, bfr[kk + 1], (short)0, a1, false, false);
            }
            accL = a0 + a1;
        }

        // ---- mask penalty + per-row chunk max ----
        const int colm = cg * 16 + l16;
        const float mmM = sMaskM[colm];
#pragma unroll
        for (int i = 0; i < 8; ++i) {
            const int row = rt * 16 + i + 8 * half;
            float x = accL[i];
            if (sMaskN[row] * mmM != 1.0f) x -= NEGV;
            accL[i] = x;
            float mx = x;
#pragma unroll
            for (int o = 1; o < 16; o <<= 1) mx = fmaxf(mx, __shfl_xor(mx, o));
            if (l16 == 0) sCMax[row * 2 + cg] = mx;
        }
        __syncthreads();

        if (t < 64) {
            const float mold = sM[t];
            const float mnew = fmaxf(mold, fmaxf(sCMax[t * 2], sCMax[t * 2 + 1]));
            const float sc   = __expf(mold - mnew);
            sM[t] = mnew; sScale[t] = sc; sL[t] *= sc;
        }
        __syncthreads();

        // ---- probs, chunk row-sums, rescale O accumulators ----
#pragma unroll
        for (int i = 0; i < 8; ++i) {
            const int row = rt * 16 + i + 8 * half;
            const float p = __expf(accL[i] - sM[row]);
            sP[row * PPITCH + colm] = f2bf(p);
            float sm = p;
#pragma unroll
            for (int o = 1; o < 16; o <<= 1) sm += __shfl_xor(sm, o);
            if (l16 == 0) sCSum[row * 2 + cg] = sm;
            const float sc = sScale[row];
#pragma unroll
            for (int j = 0; j < 8; ++j) acc[j][i] *= sc;
        }
        __syncthreads();

        if (t < 64) sL[t] += sCSum[t * 2] + sCSum[t * 2 + 1];

        // ---- stage V chunk (bf16, pre-transposed [d][m]): 256 rows x 64B ----
        {
            const unsigned short* vb = V16 + ((size_t)(b * CDIM + t)) * HW + m0;
#pragma unroll
            for (int k2 = 0; k2 < 4; ++k2)
                copy16(vb + k2 * 8, &sB[t * BVPITCH + k2 * 8]);
        }
        copy_fence();
        __syncthreads();

        // ---- O += P * V : 8 independent WMMA chains per wave ----
        {
            const v16bf pa = loadAfrag(sP + (rt * 16 + l16) * PPITCH, 0, half);
            v16bf vfr[8];
#pragma unroll
            for (int j = 0; j < 8; ++j)
                vfr[j] = loadBfrag(sB + ((cg * 8 + j) * 16 + l16) * BVPITCH + 16 * half);
#pragma unroll
            for (int j = 0; j < 8; ++j)
                acc[j] = __builtin_amdgcn_wmma_f32_16x16x32_bf16(
                    false, pa, false, vfr[j], (short)0, acc[j], false, false);
        }
        __syncthreads();
    }

    // ---- exe column (m = 4096): mask column is all-ones ----
    if (t < 64) {
        const float le   = sExe[t];
        const float mold = sM[t];
        const float mnew = fmaxf(mold, le);
        const float sc   = __expf(mold - mnew);
        const float pe   = __expf(le - mnew);
        sL[t] = sL[t] * sc + pe;
        sScale[t] = sc; sPexe[t] = pe;
    }
    __syncthreads();

    // ---- finalize: O = (O*sc + pe*V_exe) / l ; out1 = O + img ----
#pragma unroll
    for (int j = 0; j < 8; ++j) {
        const int dcol = (cg * 8 + j) * 16 + l16;
        const float vex = Vw[((size_t)b * NFULL + HW) * CDIM + dcol];
#pragma unroll
        for (int i = 0; i < 8; ++i) {
            const int row = rt * 16 + i + 8 * half;
            float o = acc[j][i] * sScale[row] + sPexe[row] * vex;
            o /= sL[row];
            const size_t gi = ((size_t)(b * CDIM + dcol)) * HW + (n0 + row);
            out1[gi] = o + img[gi];
        }
    }
}

// ----------------------- K6: gather queries from V --------------------------
__global__ __launch_bounds__(256) void query_kernel(
    const float* __restrict__ Vw, const int* __restrict__ qids,
    float* __restrict__ outq)
{
    const int b = blockIdx.x / NQ;
    const int q = blockIdx.x % NQ;
    const int d = threadIdx.x;
    const int n = qids[b * NQ + q];
    outq[((size_t)(b * NQ + q)) * CDIM + d] =
        Vw[((size_t)(b * NFULL + n)) * CDIM + d];
}

// ---------------------------------------------------------------------------
extern "C" void kernel_launch(void* const* d_in, const int* in_sizes, int n_in,
                              void* d_out, int out_size, void* d_ws, size_t ws_size,
                              hipStream_t stream) {
    (void)in_sizes; (void)n_in; (void)out_size; (void)ws_size;
    const float* img  = (const float*)d_in[0];   // (B,C,H,W)
    const float* exe  = (const float*)d_in[1];   // (B,C)
    const float* mask = (const float*)d_in[2];   // (B,H,W)
    const float* sim  = (const float*)d_in[3];   // (C,C)
    const float* vw   = (const float*)d_in[4];   // (C,C)

    float* out1 = (float*)d_out;                           // (B,C,H,W)
    float* outq = out1 + (size_t)4 * CDIM * HW;            // (B,NQ,C)

    // workspace carve-out (~59 MB)
    float* Tw     = (float*)d_ws;                          // B*HW*C f32
    float* Vw     = Tw + (size_t)4 * HW * CDIM;            // B*NFULL*C f32
    float* scores = Vw + (size_t)4 * NFULL * CDIM;         // B*HW f32
    int*   qids   = (int*)(scores + (size_t)4 * HW);       // B*NQ (pad to 1024)
    unsigned short* T16  = (unsigned short*)(qids + 1024); // B*HW*C bf16
    unsigned short* Cc16 = T16 + (size_t)4 * HW * CDIM;    // B*HW*C bf16 [m][d]
    unsigned short* V16  = Cc16 + (size_t)4 * HW * CDIM;   // B*C*HW bf16 [d][m]

    t_gemm_kernel<<<dim3(4 * 64), dim3(256), 0, stream>>>(img, sim, Tw, T16, Cc16);
    v_gemm_kernel<<<dim3(4 * 65), dim3(256), 0, stream>>>(img, exe, vw, Vw, V16);
    score_kernel <<<dim3(4 * 16), dim3(256), 0, stream>>>(Tw, exe, scores);
    topk_kernel  <<<dim3(4),      dim3(256), 0, stream>>>(scores, qids);
    attn_kernel  <<<dim3(4 * 64), dim3(256), 0, stream>>>(img, mask, T16, Cc16, V16,
                                                          Vw, scores, out1);
    query_kernel <<<dim3(4 * NQ), dim3(256), 0, stream>>>(Vw, qids, outq);
}
